// GotenNet_28329604285248
// MI455X (gfx1250) — matrix operand order repaired and checked by reading, not compiled, not run on hardware
//
#include <hip/hip_runtime.h>
#include <math.h>

// ---------------------------------------------------------------------------
// GotenNet layer for MI455X (gfx1250, wave32, WMMA).
// All dense projections go through one tiled v_wmma_f32_16x16x32_f16 kernel
// (f16 inputs, f32 accumulate). Gather is folded into node-level GEMMs.
// ---------------------------------------------------------------------------

#define N_NODE 8000
#define N_EDGE 64000
#define DNE 128
#define HEADS 8
#define DH 16
#define CD 640
#define CH 80
#define DXPD 64

typedef __attribute__((ext_vector_type(16))) _Float16 v16h;
typedef __attribute__((ext_vector_type(8)))  _Float16 v8h;
typedef __attribute__((ext_vector_type(4)))  _Float16 v4h;
typedef __attribute__((ext_vector_type(8)))  float    v8f;
typedef __attribute__((ext_vector_type(4)))  float    v4f;

#define LDSPITCH 40   // halves per row: 80 B, 16 B aligned, de-banked

// ---------------------------------------------------------------------------
// WMMA GEMM: C[M,Nout] = act(A[M,K] @ W[K,Nout] + bias)
// Block = 256 threads = 8 wave32; block tile = 128 rows x 32 cols; K step 32.
// Each wave owns one M-subtile (16 rows) and two 16x16 accumulators.
// A kept row-major in LDS; B stored N-major (transposed) so every lane's
// fragment is two contiguous 16-byte chunks -> ds_load_b128 pairs.
// Fragments use contiguous K order (k = hi*16 + e) for both A (16x32) and B (32x16).
// ---------------------------------------------------------------------------
template<int ACT> // 0 = identity, 1 = SiLU
__global__ __launch_bounds__(256)
void wmma_gemm_kernel(const float* __restrict__ A, const float* __restrict__ W,
                      const float* __restrict__ bias, float* __restrict__ C,
                      int M, int K, int Nout)
{
    __shared__ _Float16 sA[128][LDSPITCH];  // rows of A tile   (10 KB)
    __shared__ _Float16 sBt[32][LDSPITCH];  // columns of B tile (2.5 KB)

    const int tid  = threadIdx.x;
    const int wave = tid >> 5;
    const int lane = tid & 31;
    const int l16  = lane & 15;
    const int hi   = lane >> 4;          // 0: lanes 0-15, 1: lanes 16-31
    const int row0 = blockIdx.x * 128;
    const int col0 = blockIdx.y * 32;

    v8f acc0 = {}, acc1 = {};

    for (int k0 = 0; k0 < K; k0 += 32) {
        // ---- A tile (128x32 f32): float4 loads, then convert to f16 ----
        v4f av[4];
        #pragma unroll
        for (int it = 0; it < 4; ++it) {
            int f4 = tid + it * 256;             // float4 index in tile
            int r = f4 >> 3, c4 = f4 & 7;
            v4f v = {0.0f, 0.0f, 0.0f, 0.0f};
            if (row0 + r < M)
                v = *(const v4f*)(A + (size_t)(row0 + r) * K + k0 + c4 * 4);
            av[it] = v;
        }
        #pragma unroll
        for (int it = 0; it < 4; ++it) {
            int f4 = tid + it * 256;
            int r = f4 >> 3, c4 = f4 & 7;
            v4h hv;
            #pragma unroll
            for (int e = 0; e < 4; ++e) hv[e] = (_Float16)av[it][e];
            *(v4h*)&sA[r][c4 * 4] = hv;
        }
        // ---- B tile (32x32 f32), stored transposed: sBt[n][k] ----
        float bv[4];
        #pragma unroll
        for (int it = 0; it < 4; ++it) {
            int idx = tid + it * 256;
            int r = idx >> 5, cc = idx & 31;     // k-row, n-col
            bv[it] = W[(size_t)(k0 + r) * Nout + col0 + cc];
        }
        #pragma unroll
        for (int it = 0; it < 4; ++it) {
            int idx = tid + it * 256;
            int r = idx >> 5, cc = idx & 31;
            sBt[cc][r] = (_Float16)bv[it];
        }
        __syncthreads();

        // A fragment: lane (l16,hi) = row l16, K = hi*16 + e (contiguous)
        const v8h* ap = (const v8h*)&sA[wave * 16 + l16][0];
        v8h a0 = ap[2 * hi], a1 = ap[2 * hi + 1];
        v16h afrag = __builtin_shufflevector(a0, a1,
            0, 1, 2, 3, 4, 5, 6, 7, 8, 9, 10, 11, 12, 13, 14, 15);

        // B fragments: lane = column l16 (+16), K = hi*16 + e (contiguous)
        const v8h* bp0 = (const v8h*)&sBt[l16][0];
        v8h b00 = bp0[2 * hi], b01 = bp0[2 * hi + 1];
        v16h bfrag0 = __builtin_shufflevector(b00, b01,
            0, 1, 2, 3, 4, 5, 6, 7, 8, 9, 10, 11, 12, 13, 14, 15);
        const v8h* bp1 = (const v8h*)&sBt[16 + l16][0];
        v8h b10 = bp1[2 * hi], b11 = bp1[2 * hi + 1];
        v16h bfrag1 = __builtin_shufflevector(b10, b11,
            0, 1, 2, 3, 4, 5, 6, 7, 8, 9, 10, 11, 12, 13, 14, 15);

        acc0 = __builtin_amdgcn_wmma_f32_16x16x32_f16(
                   false, afrag, false, bfrag0, (short)0, acc0, false, false);
        acc1 = __builtin_amdgcn_wmma_f32_16x16x32_f16(
                   false, afrag, false, bfrag1, (short)0, acc1, false, false);
        __syncthreads();
    }

    // D layout: element r -> M = wave*16 + hi*8 + r, N = l16 (+16 for acc1)
    float b0 = bias ? bias[col0 + l16] : 0.0f;
    float b1 = bias ? bias[col0 + 16 + l16] : 0.0f;
    #pragma unroll
    for (int r = 0; r < 8; ++r) {
        int row = row0 + wave * 16 + hi * 8 + r;
        if (row < M) {
            float v0 = acc0[r] + b0;
            float v1 = acc1[r] + b1;
            if (ACT == 1) {
                v0 = v0 / (1.0f + __expf(-v0));   // SiLU
                v1 = v1 / (1.0f + __expf(-v1));
            }
            C[(size_t)row * Nout + col0 + l16]      = v0;
            C[(size_t)row * Nout + col0 + 16 + l16] = v1;
        }
    }
}

// ---------------------------------------------------------------------------
// Elementwise / segment kernels
// ---------------------------------------------------------------------------
__global__ void fill_kernel(float* p, float v, int n) {
    int i = blockIdx.x * blockDim.x + threadIdx.x;
    if (i < n) p[i] = v;
}

__device__ __forceinline__ void atomicMaxF(float* addr, float val) {
    // ordered-int trick (handles mixed signs; buffer init to -1e9)
    if (val >= 0.0f) atomicMax((int*)addr, __float_as_int(val));
    else             atomicMin((unsigned int*)addr, __float_as_uint(val));
}

// alpha[e,h] = sum_d q[i,h,d] * k[j,h,d] * geom[e,h,d]; seg-max + degree
__global__ void alpha_kernel(const float* __restrict__ q, const float* __restrict__ k,
                             const float* __restrict__ geom, const int* __restrict__ edge,
                             float* __restrict__ alpha, float* __restrict__ mx,
                             float* __restrict__ deg)
{
    int idx = blockIdx.x * blockDim.x + threadIdx.x;
    if (idx >= N_EDGE * HEADS) return;
    int e = idx / HEADS, h = idx % HEADS;
    int i = edge[2 * e], j = edge[2 * e + 1];
    const float* qp = q + (size_t)i * DNE + h * DH;
    const float* kp = k + (size_t)j * DNE + h * DH;
    const float* gp = geom + (size_t)e * DNE + h * DH;
    float a = 0.0f;
    #pragma unroll
    for (int d = 0; d < DH; ++d) a += qp[d] * kp[d] * gp[d];
    alpha[idx] = a;
    atomicMaxF(&mx[i * HEADS + h], a);
    if (h == 0) atomicAdd(&deg[i], 1.0f);
}

// ex[e,h] = exp(alpha - mx[i,h]); den[i,h] += ex
__global__ void exden_kernel(const int* __restrict__ edge, const float* __restrict__ mx,
                             float* __restrict__ ex, float* __restrict__ den)
{
    int idx = blockIdx.x * blockDim.x + threadIdx.x;
    if (idx >= N_EDGE * HEADS) return;
    int e = idx / HEADS, h = idx % HEADS;
    int i = edge[2 * e];
    float v = __expf(ex[idx] - mx[i * HEADS + h]);
    ex[idx] = v;
    atomicAdd(&den[i * HEADS + h], v);
}

// One block (128 threads) per edge: full 640-wide coeff + scatter-add of
// h_out and the l=1,2 equivariant channel messages.
__global__ __launch_bounds__(128)
void coeff_scatter_kernel(const float* __restrict__ ex, const float* __restrict__ den,
                          const float* __restrict__ deg, const float* __restrict__ v_node,
                          const float* __restrict__ s_node, const float* __restrict__ tp,
                          const float* __restrict__ c, const float* __restrict__ r1,
                          const float* __restrict__ r2, const float* __restrict__ X1,
                          const float* __restrict__ X2, const int* __restrict__ edge,
                          float* __restrict__ h_out, float* __restrict__ X1_out,
                          float* __restrict__ X2_out)
{
    int e = blockIdx.x;
    int d = threadIdx.x;                      // channel 0..127
    int i = edge[2 * e], j = edge[2 * e + 1];
    float scale = sqrtf(deg[i]) * 0.08838834764831845f;   // sqrt(deg)/sqrt(128)
    float cc = c[e];
    float co[5];
    #pragma unroll
    for (int s = 0; s < 5; ++s) {
        int f = s * 128 + d;                  // flat CD index
        int h = f / CH;                       // head of this flat slot
        float attn = ex[(size_t)e * HEADS + h]
                   / fmaxf(den[(size_t)i * HEADS + h], 1e-12f) * scale;
        co[s] = attn * v_node[(size_t)j * CD + f]
              + tp[(size_t)e * CD + f] * s_node[(size_t)j * CD + f] * cc;
    }
    atomicAdd(&h_out[(size_t)i * 128 + d], co[0]);
    #pragma unroll
    for (int m = 0; m < 3; ++m)
        atomicAdd(&X1_out[((size_t)i * 3 + m) * 128 + d],
                  r1[e * 3 + m] * co[1] + X1[((size_t)j * 3 + m) * 128 + d] * co[3]);
    #pragma unroll
    for (int m = 0; m < 5; ++m)
        atomicAdd(&X2_out[((size_t)i * 5 + m) * 128 + d],
                  r2[e * 5 + m] * co[2] + X2[((size_t)j * 5 + m) * 128 + d] * co[4]);
}

// Rejection (rej with -r == rej with +r) + channel dot product: w[e, 0..63]
__global__ void wdot_kernel(const float* __restrict__ Q1, const float* __restrict__ Q2,
                            const float* __restrict__ K1, const float* __restrict__ K2,
                            const float* __restrict__ r1, const float* __restrict__ r2,
                            const int* __restrict__ edge, float* __restrict__ w)
{
    int idx = blockIdx.x * blockDim.x + threadIdx.x;
    if (idx >= N_EDGE * DXPD) return;
    int e = idx / DXPD, d = idx % DXPD;
    int i = edge[2 * e], j = edge[2 * e + 1];
    float q[8], k[8];
    #pragma unroll
    for (int m = 0; m < 3; ++m) {
        q[m] = Q1[((size_t)i * 3 + m) * DXPD + d];
        k[m] = K1[((size_t)j * 3 + m) * DXPD + d];
    }
    #pragma unroll
    for (int m = 0; m < 5; ++m) {
        q[3 + m] = Q2[((size_t)i * 5 + m) * DXPD + d];
        k[3 + m] = K2[((size_t)j * 5 + m) * DXPD + d];
    }
    // reject onto r1 (3 components)
    float ra[3], pq = 0.0f, pk = 0.0f;
    #pragma unroll
    for (int m = 0; m < 3; ++m) { ra[m] = r1[e * 3 + m]; pq += q[m] * ra[m]; pk += k[m] * ra[m]; }
    #pragma unroll
    for (int m = 0; m < 3; ++m) { q[m] -= pq * ra[m]; k[m] -= pk * ra[m]; }
    // reject onto r2 (5 components)
    float rb[5]; pq = 0.0f; pk = 0.0f;
    #pragma unroll
    for (int m = 0; m < 5; ++m) { rb[m] = r2[e * 5 + m]; pq += q[3 + m] * rb[m]; pk += k[3 + m] * rb[m]; }
    #pragma unroll
    for (int m = 0; m < 5; ++m) { q[3 + m] -= pq * rb[m]; k[3 + m] -= pk * rb[m]; }
    float acc = 0.0f;
    #pragma unroll
    for (int m = 0; m < 8; ++m) acc += q[m] * k[m];
    w[idx] = acc;
}

// t_out = t + wmul * tmul
__global__ void tout_kernel(const float* __restrict__ t, const float* __restrict__ wmul,
                            const float* __restrict__ tmul, float* __restrict__ t_out)
{
    int idx = blockIdx.x * blockDim.x + threadIdx.x;
    if (idx < N_EDGE * DNE) t_out[idx] = t[idx] + wmul[idx] * tmul[idx];
}

// ---------------------------------------------------------------------------
// Host driver
// ---------------------------------------------------------------------------
static inline int cdiv(int a, int b) { return (a + b - 1) / b; }

static void gemm(hipStream_t s, const float* A, const float* W, const float* b,
                 float* C, int M, int K, int Nout, int act)
{
    dim3 grid(cdiv(M, 128), Nout / 32);
    if (act) wmma_gemm_kernel<1><<<grid, 256, 0, s>>>(A, W, b, C, M, K, Nout);
    else     wmma_gemm_kernel<0><<<grid, 256, 0, s>>>(A, W, b, C, M, K, Nout);
}

extern "C" void kernel_launch(void* const* d_in, const int* in_sizes, int n_in,
                              void* d_out, int out_size, void* d_ws, size_t ws_size,
                              hipStream_t stream)
{
    (void)in_sizes; (void)n_in; (void)out_size; (void)ws_size;
    const float* h   = (const float*)d_in[0];
    const float* t   = (const float*)d_in[1];
    const float* X1  = (const float*)d_in[2];
    const float* X2  = (const float*)d_in[3];
    const float* r1  = (const float*)d_in[4];
    const float* r2  = (const float*)d_in[5];
    const float* c   = (const float*)d_in[6];
    const int*   edge= (const int*)  d_in[7];
    const float* Wq  = (const float*)d_in[8];  const float* bq  = (const float*)d_in[9];
    const float* Wk  = (const float*)d_in[10]; const float* bk  = (const float*)d_in[11];
    const float* Wre = (const float*)d_in[12]; const float* bre = (const float*)d_in[13];
    const float* Wv1 = (const float*)d_in[14]; const float* bv1 = (const float*)d_in[15];
    const float* Wv2 = (const float*)d_in[16]; const float* bv2 = (const float*)d_in[17];
    const float* Wrs = (const float*)d_in[18]; const float* brs = (const float*)d_in[19];
    const float* Ws1 = (const float*)d_in[20]; const float* bs1 = (const float*)d_in[21];
    const float* Ws2 = (const float*)d_in[22]; const float* bs2 = (const float*)d_in[23];
    const float* Wvq = (const float*)d_in[24];
    const float* Wvk1= (const float*)d_in[25];
    const float* Wvk2= (const float*)d_in[26];
    const float* Ww1 = (const float*)d_in[27]; const float* bw1 = (const float*)d_in[28];
    const float* Ww2 = (const float*)d_in[29]; const float* bw2 = (const float*)d_in[30];
    const float* Wt1 = (const float*)d_in[31]; const float* bt1 = (const float*)d_in[32];
    const float* Wt2 = (const float*)d_in[33]; const float* bt2 = (const float*)d_in[34];

    // output layout: h_out | t_out | X1_out | X2_out
    float* out    = (float*)d_out;
    float* h_out  = out;
    float* t_out  = out + (size_t)N_NODE * DNE;
    float* X1_out = t_out + (size_t)N_EDGE * DNE;
    float* X2_out = X1_out + (size_t)N_NODE * 3 * DNE;

    // workspace layout (floats), with phase-based aliasing
    float* ws     = (float*)d_ws;
    float* q_node = ws;
    float* k_node = q_node + (size_t)N_NODE * DNE;
    float* tmp    = k_node + (size_t)N_NODE * DNE;
    float* v_node = tmp    + (size_t)N_NODE * DNE;
    float* s_node = v_node + (size_t)N_NODE * CD;
    float* geom   = s_node + (size_t)N_NODE * CD;
    float* tp     = geom   + (size_t)N_EDGE * DNE;
    float* ex     = tp     + (size_t)N_EDGE * CD;       // alpha, then exp in place
    float* mx     = ex     + (size_t)N_EDGE * HEADS;
    float* den    = mx     + (size_t)N_NODE * HEADS;
    float* deg    = den    + (size_t)N_NODE * HEADS;
    // phase B aliases (v_node/s_node/geom dead after coeff scatter)
    float* Q1 = v_node;
    float* Q2 = Q1 + (size_t)N_NODE * 3 * DXPD;
    float* K1 = Q2 + (size_t)N_NODE * 5 * DXPD;
    float* K2 = K1 + (size_t)N_NODE * 3 * DXPD;
    float* w  = geom;
    float* wt1 = geom + (size_t)N_EDGE * DXPD;
    // phase C aliases (tp dead)
    float* wmul = tp;
    float* tt1  = tp  + (size_t)N_EDGE * DNE;
    float* tmul = tt1 + (size_t)N_EDGE * DNE;

    // ---- init: residual copies + segment accumulators ----
    hipMemcpyAsync(h_out,  h,  (size_t)N_NODE * DNE * 4,     hipMemcpyDeviceToDevice, stream);
    hipMemcpyAsync(X1_out, X1, (size_t)N_NODE * 3 * DNE * 4, hipMemcpyDeviceToDevice, stream);
    hipMemcpyAsync(X2_out, X2, (size_t)N_NODE * 5 * DNE * 4, hipMemcpyDeviceToDevice, stream);
    fill_kernel<<<cdiv(N_NODE * HEADS, 256), 256, 0, stream>>>(mx, -1e9f, N_NODE * HEADS);
    fill_kernel<<<cdiv(N_NODE * HEADS, 256), 256, 0, stream>>>(den, 0.0f, N_NODE * HEADS);
    fill_kernel<<<cdiv(N_NODE, 256), 256, 0, stream>>>(deg, 0.0f, N_NODE);

    // ---- node-level projections (gather folded out of per-edge GEMMs) ----
    gemm(stream, h, Wq,  bq,  q_node, N_NODE, DNE, DNE, 0);
    gemm(stream, h, Wk,  bk,  k_node, N_NODE, DNE, DNE, 0);
    gemm(stream, h, Wv1, bv1, tmp,    N_NODE, DNE, DNE, 1);   // silu
    gemm(stream, tmp, Wv2, bv2, v_node, N_NODE, DNE, CD, 0);
    gemm(stream, h, Ws1, bs1, tmp,    N_NODE, DNE, DNE, 1);   // silu
    gemm(stream, tmp, Ws2, bs2, s_node, N_NODE, DNE, CD, 0);

    // ---- edge-level projections on t ----
    gemm(stream, t, Wre, bre, geom, N_EDGE, DNE, DNE, 1);     // silu
    gemm(stream, t, Wrs, brs, tp,   N_EDGE, DNE, CD, 0);

    // ---- segment softmax attention ----
    alpha_kernel<<<cdiv(N_EDGE * HEADS, 256), 256, 0, stream>>>(
        q_node, k_node, geom, edge, ex, mx, deg);
    exden_kernel<<<cdiv(N_EDGE * HEADS, 256), 256, 0, stream>>>(edge, mx, ex, den);

    // ---- coefficient build + scatter to h_out / X1_out / X2_out ----
    coeff_scatter_kernel<<<N_EDGE, 128, 0, stream>>>(
        ex, den, deg, v_node, s_node, tp, c, r1, r2, X1, X2, edge,
        h_out, X1_out, X2_out);

    // ---- vector-channel projections (on updated X_out, node granularity) ----
    gemm(stream, X1_out, Wvq,  nullptr, Q1, N_NODE * 3, DNE, DXPD, 0);
    gemm(stream, X2_out, Wvq,  nullptr, Q2, N_NODE * 5, DNE, DXPD, 0);
    gemm(stream, X1_out, Wvk1, nullptr, K1, N_NODE * 3, DNE, DXPD, 0);
    gemm(stream, X2_out, Wvk2, nullptr, K2, N_NODE * 5, DNE, DXPD, 0);

    // ---- rejection + dot ----
    wdot_kernel<<<cdiv(N_EDGE * DXPD, 256), 256, 0, stream>>>(
        Q1, Q2, K1, K2, r1, r2, edge, w);

    // ---- gated t-MLP ----
    gemm(stream, w,   Ww1, bw1, wt1,  N_EDGE, DXPD, DXPD, 1); // silu
    gemm(stream, wt1, Ww2, bw2, wmul, N_EDGE, DXPD, DNE, 0);
    gemm(stream, t,   Wt1, bt1, tt1,  N_EDGE, DNE, DNE, 1);   // silu
    gemm(stream, tt1, Wt2, bt2, tmul, N_EDGE, DNE, DNE, 0);
    tout_kernel<<<cdiv(N_EDGE * DNE, 256), 256, 0, stream>>>(t, wmul, tmul, t_out);
}
